// ParallelAttention_32839319945791
// MI455X (gfx1250) — matrix operand order repaired and checked by reading, not compiled, not run on hardware
//
#include <hip/hip_runtime.h>

typedef __bf16 bf16_t;
typedef __attribute__((ext_vector_type(16))) __bf16 v16bf;
typedef __attribute__((ext_vector_type(8)))  __bf16 v8bf;
typedef __attribute__((ext_vector_type(8)))  float  v8f;

#define B_ 32
#define C_ 128
#define V_ 64
#define T_ 128
#define C8_ 16
#define C2_ 64
#define CR_ 8

// ---------------------------------------------------------------------------
// x (f32) -> bf16 copy, fused with per-(b,c) mean for the SE branch.
// ---------------------------------------------------------------------------
__global__ __launch_bounds__(256) void cvt_reduce_kernel(
    const float* __restrict__ x, bf16_t* __restrict__ xb, float* __restrict__ meanOut)
{
  int bc = blockIdx.x;
  size_t base = (size_t)bc * (V_ * T_);
  __shared__ float red[256];
  float acc = 0.f;
  for (int i = threadIdx.x; i < V_ * T_; i += 256) {
    float v = x[base + i];
    xb[base + i] = (bf16_t)v;
    acc += v;
  }
  red[threadIdx.x] = acc;
  __syncthreads();
  for (int off = 128; off > 0; off >>= 1) {
    if (threadIdx.x < off) red[threadIdx.x] += red[threadIdx.x + off];
    __syncthreads();
  }
  if (threadIdx.x == 0) meanOut[bc] = red[0] * (1.f / (V_ * T_));
}

// Weight f32 [CO][CIN][3][3] -> bf16 [CO][p][CIN]  (p = kh*3+kw, K-major = p*CIN+ci)
__global__ void cvt_weight_kernel(const float* __restrict__ src, bf16_t* __restrict__ dst,
                                  int cin, int n)
{
  int i = blockIdx.x * 256 + threadIdx.x;
  if (i < n) {
    int K = cin * 9;
    int co = i / K;
    int r = i - co * K;
    int p = r / cin;
    int ci = r - p * cin;
    dst[i] = (bf16_t)src[co * K + ci * 9 + p];
  }
}

// ---------------------------------------------------------------------------
// SE block: scale[b,c] = sigmoid( relu(mean[b,:] @ w1^T) @ w2^T )[c]
// ---------------------------------------------------------------------------
__global__ __launch_bounds__(128) void se_kernel(
    const float* __restrict__ mean, const float* __restrict__ w1,
    const float* __restrict__ w2, float* __restrict__ scale)
{
  int b = blockIdx.x;
  int tid = threadIdx.x;
  __shared__ float sm[C_];
  __shared__ float sh[CR_];
  sm[tid] = mean[b * C_ + tid];
  __syncthreads();
  if (tid < CR_) {
    float a = 0.f;
    for (int c = 0; c < C_; ++c) a += sm[c] * w1[tid * C_ + c];
    sh[tid] = fmaxf(a, 0.f);
  }
  __syncthreads();
  float a = 0.f;
  for (int r = 0; r < CR_; ++r) a += sh[r] * w2[tid * CR_ + r];
  scale[b * C_ + tid] = 1.f / (1.f + __expf(-a));
}

// ---------------------------------------------------------------------------
// Implicit-GEMM 3x3 conv via v_wmma_f32_16x16x32_bf16.
// K-order is kernel-position-major: k = p*CIN + ci. Two independent
// accumulator chains (alternating k-steps) keep the XDL pipe busy despite the
// WMMA->WMMA RAW latency; summed at the end.
// ---------------------------------------------------------------------------
template <int CIN, int COUT, bool TANH>
__global__ __launch_bounds__(128) void conv3_bf16_kernel(
    const bf16_t* __restrict__ xin, const bf16_t* __restrict__ wgt, bf16_t* __restrict__ out)
{
  constexpr int K = CIN * 9;
  constexpr int TN = 64;
  constexpr int TLW = TN + 2;
  constexpr int CINP = CIN + 8;
  __shared__ __attribute__((aligned(32))) bf16_t sX[3 * TLW * CINP];

  const int tTiles = T_ / TN; // 2
  int gx = blockIdx.x;
  int tTile = gx % tTiles;
  int coutBase = (gx / tTiles) * 16;
  int v = blockIdx.y;
  int b = blockIdx.z;
  int t0 = tTile * TN;
  int tid = threadIdx.x;

  for (int idx = tid; idx < 3 * CIN * TLW; idx += 128) {
    int tl = idx % TLW;
    int rem = idx / TLW;
    int ci = rem % CIN;
    int kh = rem / CIN;
    int row = v + kh - 1;
    int t = t0 + tl - 1;
    bf16_t val = (bf16_t)0.f;
    if ((unsigned)row < (unsigned)V_ && (unsigned)t < (unsigned)T_)
      val = xin[(((size_t)b * CIN + ci) * V_ + row) * T_ + t];
    sX[(kh * TLW + tl) * CINP + ci] = val;
  }
  __syncthreads();

  int wave = tid >> 5;
  int lane = tid & 31;
  int hi = lane >> 4;
  int mrow = lane & 15;
  int ncol = lane & 15;
  int tl_base = wave * 16 + ncol;

  v8f acc0 = {}, acc1 = {};
  const bf16_t* wrow = wgt + (size_t)(coutBase + mrow) * K;

  for (int p = 0; p < 9; ++p) {
    int kh = p / 3, kw = p - (p / 3) * 3; // uniform
    const bf16_t* xrow = sX + (kh * TLW + tl_base + kw) * CINP;
    const bf16_t* wp = wrow + p * CIN;
#pragma unroll
    for (int ci0 = 0; ci0 < CIN; ci0 += 32) {
      union { v16bf v; v8bf h[2]; } A, Bm;
      A.h[0] = *(const v8bf*)(wp + ci0 + hi * 8);
      A.h[1] = *(const v8bf*)(wp + ci0 + 16 + hi * 8);
      Bm.h[0] = *(const v8bf*)(xrow + ci0 + hi * 8);
      Bm.h[1] = *(const v8bf*)(xrow + ci0 + 16 + hi * 8);
      if (((ci0 >> 5) & 1) == 0)
        acc0 = __builtin_amdgcn_wmma_f32_16x16x32_bf16(false, A.v, false, Bm.v, (short)0, acc0,
                                                       false, false);
      else
        acc1 = __builtin_amdgcn_wmma_f32_16x16x32_bf16(false, A.v, false, Bm.v, (short)0, acc1,
                                                       false, false);
    }
  }

#pragma unroll
  for (int j = 0; j < 8; ++j) {
    int co = coutBase + j + hi * 8;
    float val = acc0[j] + acc1[j];
    if (TANH) val = tanhf(val);
    int t = t0 + wave * 16 + ncol;
    out[(((size_t)b * COUT + co) * V_ + v) * T_ + t] = (bf16_t)val;
  }
}

// Output-projection conv: bf16 in, f32 out scaled by sigma; write or accumulate.
template <int CIN, int COUT, bool ACCUM>
__global__ __launch_bounds__(128) void conv3_f32_kernel(
    const bf16_t* __restrict__ xin, const bf16_t* __restrict__ wgt,
    const float* __restrict__ sigma, float* __restrict__ out)
{
  constexpr int K = CIN * 9;
  constexpr int TN = 64;
  constexpr int TLW = TN + 2;
  constexpr int CINP = CIN + 8;
  __shared__ __attribute__((aligned(32))) bf16_t sX[3 * TLW * CINP];

  const int tTiles = T_ / TN;
  int gx = blockIdx.x;
  int tTile = gx % tTiles;
  int coutBase = (gx / tTiles) * 16;
  int v = blockIdx.y;
  int b = blockIdx.z;
  int t0 = tTile * TN;
  int tid = threadIdx.x;

  for (int idx = tid; idx < 3 * CIN * TLW; idx += 128) {
    int tl = idx % TLW;
    int rem = idx / TLW;
    int ci = rem % CIN;
    int kh = rem / CIN;
    int row = v + kh - 1;
    int t = t0 + tl - 1;
    bf16_t val = (bf16_t)0.f;
    if ((unsigned)row < (unsigned)V_ && (unsigned)t < (unsigned)T_)
      val = xin[(((size_t)b * CIN + ci) * V_ + row) * T_ + t];
    sX[(kh * TLW + tl) * CINP + ci] = val;
  }
  __syncthreads();

  int wave = tid >> 5;
  int lane = tid & 31;
  int hi = lane >> 4;
  int mrow = lane & 15;
  int ncol = lane & 15;
  int tl_base = wave * 16 + ncol;

  v8f acc0 = {}, acc1 = {};
  const bf16_t* wrow = wgt + (size_t)(coutBase + mrow) * K;

  for (int p = 0; p < 9; ++p) {
    int kh = p / 3, kw = p - (p / 3) * 3;
    const bf16_t* xrow = sX + (kh * TLW + tl_base + kw) * CINP;
    const bf16_t* wp = wrow + p * CIN;
#pragma unroll
    for (int ci0 = 0; ci0 < CIN; ci0 += 32) {
      union { v16bf v; v8bf h[2]; } A, Bm;
      A.h[0] = *(const v8bf*)(wp + ci0 + hi * 8);
      A.h[1] = *(const v8bf*)(wp + ci0 + 16 + hi * 8);
      Bm.h[0] = *(const v8bf*)(xrow + ci0 + hi * 8);
      Bm.h[1] = *(const v8bf*)(xrow + ci0 + 16 + hi * 8);
      if (((ci0 >> 5) & 1) == 0)
        acc0 = __builtin_amdgcn_wmma_f32_16x16x32_bf16(false, A.v, false, Bm.v, (short)0, acc0,
                                                       false, false);
      else
        acc1 = __builtin_amdgcn_wmma_f32_16x16x32_bf16(false, A.v, false, Bm.v, (short)0, acc1,
                                                       false, false);
    }
  }

  float s = sigma[0];
#pragma unroll
  for (int j = 0; j < 8; ++j) {
    int co = coutBase + j + hi * 8;
    int t = t0 + wave * 16 + ncol;
    size_t o = (((size_t)b * COUT + co) * V_ + v) * T_ + t;
    float val = (acc0[j] + acc1[j]) * s;
    if (ACCUM) out[o] += val; else out[o] = val;
  }
}

// ---------------------------------------------------------------------------
// V-attention: per (b,t) block.
// ---------------------------------------------------------------------------
__global__ __launch_bounds__(128) void vattn_kernel(
    const bf16_t* __restrict__ qb, const bf16_t* __restrict__ kb,
    const bf16_t* __restrict__ vb, bf16_t* __restrict__ avb)
{
  constexpr int SQP = 40;
  constexpr int SVP = 72;
  constexpr int SAP = 72;
  constexpr int SLP = V_ + 1;
  int t = blockIdx.x;
  int b = blockIdx.y;
  __shared__ __attribute__((aligned(32))) bf16_t sQt[V_ * SQP];   // [v][c], c padded to 32
  __shared__ __attribute__((aligned(32))) bf16_t sKt[V_ * SQP];   // [w][c]
  __shared__ __attribute__((aligned(32))) bf16_t sV[C2_ * SVP];   // [c][w]
  __shared__ __attribute__((aligned(32))) bf16_t sAttn[V_ * SAP]; // [v][w]
  __shared__ float sL[V_ * SLP];                                  // [v][w]
  __shared__ float sRed[128];
  int tid = threadIdx.x;

  for (int i = tid; i < V_ * C8_; i += 128) {
    int vv = i & 63, c = i >> 6;
    sQt[vv * SQP + c] = qb[(((size_t)b * C8_ + c) * V_ + vv) * T_ + t];
    sKt[vv * SQP + c] = kb[(((size_t)b * C8_ + c) * V_ + vv) * T_ + t];
  }
  for (int i = tid; i < V_ * 16; i += 128) {
    int vv = i >> 4, c = 16 + (i & 15);
    sQt[vv * SQP + c] = (bf16_t)0.f;
    sKt[vv * SQP + c] = (bf16_t)0.f;
  }
  for (int i = tid; i < C2_ * V_; i += 128) {
    int w = i & 63, c = i >> 6;
    sV[c * SVP + w] = vb[(((size_t)b * C2_ + c) * V_ + w) * T_ + t];
  }
  __syncthreads();

  int wave = tid >> 5, lane = tid & 31, hi = lane >> 4, nl = lane & 15;

  { // logits: M=v (tile=wave), N=w (4 tiles), K=32 (zero-padded)
    union { v16bf v; v8bf h[2]; } A;
    int m = wave * 16 + nl;
    A.h[0] = *(const v8bf*)(&sQt[m * SQP + hi * 8]);
    A.h[1] = *(const v8bf*)(&sQt[m * SQP + 16 + hi * 8]);
    for (int nt = 0; nt < 4; ++nt) {
      int n = nt * 16 + nl;
      union { v16bf v; v8bf h[2]; } Bm;
      Bm.h[0] = *(const v8bf*)(&sKt[n * SQP + hi * 8]);
      Bm.h[1] = *(const v8bf*)(&sKt[n * SQP + 16 + hi * 8]);
      v8f acc = {};
      acc = __builtin_amdgcn_wmma_f32_16x16x32_bf16(false, A.v, false, Bm.v, (short)0, acc,
                                                    false, false);
#pragma unroll
      for (int j = 0; j < 8; ++j) sL[(wave * 16 + j + hi * 8) * SLP + n] = acc[j];
    }
  }
  __syncthreads();

  { // softmax over w: 2 threads per row
    int row = tid >> 1, half = tid & 1;
    int w0 = half * 32;
    float mx = -1e30f;
    for (int w = w0; w < w0 + 32; ++w) mx = fmaxf(mx, sL[row * SLP + w]);
    sRed[tid] = mx;
    __syncthreads();
    mx = fmaxf(sRed[row * 2], sRed[row * 2 + 1]);
    float sum = 0.f;
    for (int w = w0; w < w0 + 32; ++w) {
      float e = __expf(sL[row * SLP + w] - mx);
      sL[row * SLP + w] = e;
      sum += e;
    }
    __syncthreads();
    sRed[tid] = sum;
    __syncthreads();
    float inv = 1.f / (sRed[row * 2] + sRed[row * 2 + 1]);
    for (int w = w0; w < w0 + 32; ++w)
      sAttn[row * SAP + w] = (bf16_t)(sL[row * SLP + w] * inv);
  }
  __syncthreads();

  { // av: M=v (tile=wave), N=c (4 tiles), K=w=64
    int m = wave * 16 + nl;
    for (int nt = 0; nt < 4; ++nt) {
      int n = nt * 16 + nl;
      v8f acc = {};
#pragma unroll
      for (int kb0 = 0; kb0 < V_; kb0 += 32) {
        union { v16bf v; v8bf h[2]; } A, Bm;
        A.h[0] = *(const v8bf*)(&sAttn[m * SAP + kb0 + hi * 8]);
        A.h[1] = *(const v8bf*)(&sAttn[m * SAP + kb0 + 16 + hi * 8]);
        Bm.h[0] = *(const v8bf*)(&sV[n * SVP + kb0 + hi * 8]);
        Bm.h[1] = *(const v8bf*)(&sV[n * SVP + kb0 + 16 + hi * 8]);
        acc = __builtin_amdgcn_wmma_f32_16x16x32_bf16(false, A.v, false, Bm.v, (short)0, acc,
                                                      false, false);
      }
#pragma unroll
      for (int j = 0; j < 8; ++j) {
        int vv = wave * 16 + j + hi * 8;
        avb[(((size_t)b * C2_ + n) * V_ + vv) * T_ + t] = (bf16_t)acc[j];
      }
    }
  }
}

// ---------------------------------------------------------------------------
// T-attention: per (b,v, 32-row t-chunk) block. V rows staged with
// global_load_async_to_lds_b32 (ASYNCcnt path; rows are contiguous in s).
// ---------------------------------------------------------------------------
__global__ __launch_bounds__(128) void tattn_kernel(
    const bf16_t* __restrict__ qb, const bf16_t* __restrict__ kb,
    const bf16_t* __restrict__ vb, bf16_t* __restrict__ avb)
{
  constexpr int SQP = 40;
  constexpr int SVP = T_ + 8;   // 136
  constexpr int SAP = T_ + 8;   // 136
  constexpr int SLP = T_ + 1;   // 129
  int tc = blockIdx.x, v = blockIdx.y, b = blockIdx.z;
  int tbase = tc * 32;
  __shared__ __attribute__((aligned(32))) bf16_t sQt[32 * SQP];   // [t-local][c]
  __shared__ __attribute__((aligned(32))) bf16_t sKt[T_ * SQP];   // [s][c]
  __shared__ __attribute__((aligned(32))) bf16_t sV[C2_ * SVP];   // [c][s]
  __shared__ __attribute__((aligned(32))) bf16_t sAttn[32 * SAP]; // [t][s]
  __shared__ float sL[32 * SLP];                                  // [t][s]
  __shared__ float sRed[128];
  int tid = threadIdx.x;

  // async global->LDS stage of V: 64 rows x 64 dwords, row-contiguous
  for (int i = tid; i < C2_ * (T_ / 2); i += 128) {
    int c = i >> 6;       // 64 dwords per row
    int sd = i & 63;      // dword index within row
    const bf16_t* g = vb + (((size_t)b * C2_ + c) * V_ + v) * T_ + sd * 2;
    unsigned lds = (unsigned)(size_t)(&sV[c * SVP + sd * 2]);
    asm volatile("global_load_async_to_lds_b32 %0, %1, off"
                 :: "v"(lds), "v"(g) : "memory");
  }

  for (int i = tid; i < 32 * C8_; i += 128) {
    int tl = i & 31, c = i >> 5;
    sQt[tl * SQP + c] = qb[(((size_t)b * C8_ + c) * V_ + v) * T_ + tbase + tl];
  }
  for (int i = tid; i < 32 * 16; i += 128) {
    int tl = i >> 4, c = 16 + (i & 15);
    sQt[tl * SQP + c] = (bf16_t)0.f;
  }
  for (int i = tid; i < T_ * C8_; i += 128) {
    int s = i & 127, c = i >> 7;
    sKt[s * SQP + c] = kb[(((size_t)b * C8_ + c) * V_ + v) * T_ + s];
  }
  for (int i = tid; i < T_ * 16; i += 128) {
    int s = i >> 4, c = 16 + (i & 15);
    sKt[s * SQP + c] = (bf16_t)0.f;
  }
  asm volatile("s_wait_asynccnt 0" ::: "memory");
  __syncthreads();

  int wave = tid >> 5, lane = tid & 31, hi = lane >> 4, nl = lane & 15;

  { // logits: M=32 (2 m-tiles), N=128 (8 n-tiles)
    int mtile = wave & 1;
    int m = mtile * 16 + nl;
    union { v16bf v; v8bf h[2]; } A;
    A.h[0] = *(const v8bf*)(&sQt[m * SQP + hi * 8]);
    A.h[1] = *(const v8bf*)(&sQt[m * SQP + 16 + hi * 8]);
    int nt0 = (wave >> 1) * 4;
    for (int ntI = 0; ntI < 4; ++ntI) {
      int n = (nt0 + ntI) * 16 + nl;
      union { v16bf v; v8bf h[2]; } Bm;
      Bm.h[0] = *(const v8bf*)(&sKt[n * SQP + hi * 8]);
      Bm.h[1] = *(const v8bf*)(&sKt[n * SQP + 16 + hi * 8]);
      v8f acc = {};
      acc = __builtin_amdgcn_wmma_f32_16x16x32_bf16(false, A.v, false, Bm.v, (short)0, acc,
                                                    false, false);
#pragma unroll
      for (int j = 0; j < 8; ++j) sL[(mtile * 16 + j + hi * 8) * SLP + n] = acc[j];
    }
  }
  __syncthreads();

  { // softmax over s: 4 threads per row
    int row = tid >> 2, q = tid & 3;
    int s0 = q * 32;
    float mx = -1e30f;
    for (int s = s0; s < s0 + 32; ++s) mx = fmaxf(mx, sL[row * SLP + s]);
    sRed[tid] = mx;
    __syncthreads();
    mx = fmaxf(fmaxf(sRed[row * 4], sRed[row * 4 + 1]),
               fmaxf(sRed[row * 4 + 2], sRed[row * 4 + 3]));
    float sum = 0.f;
    for (int s = s0; s < s0 + 32; ++s) {
      float e = __expf(sL[row * SLP + s] - mx);
      sL[row * SLP + s] = e;
      sum += e;
    }
    __syncthreads();
    sRed[tid] = sum;
    __syncthreads();
    float inv = 1.f / (sRed[row * 4] + sRed[row * 4 + 1] + sRed[row * 4 + 2] + sRed[row * 4 + 3]);
    for (int s = s0; s < s0 + 32; ++s)
      sAttn[row * SAP + s] = (bf16_t)(sL[row * SLP + s] * inv);
  }
  __syncthreads();

  { // av: M=32 t (2 m-tiles), N=64 c (4 n-tiles), K=128 s
    int mtile = wave & 1;
    int m = mtile * 16 + nl;
    for (int ntI = 0; ntI < 2; ++ntI) {
      int nt = (wave >> 1) * 2 + ntI;
      int n = nt * 16 + nl;
      v8f acc = {};
#pragma unroll
      for (int kb0 = 0; kb0 < T_; kb0 += 32) {
        union { v16bf v; v8bf h[2]; } A, Bm;
        A.h[0] = *(const v8bf*)(&sAttn[m * SAP + kb0 + hi * 8]);
        A.h[1] = *(const v8bf*)(&sAttn[m * SAP + kb0 + 16 + hi * 8]);
        Bm.h[0] = *(const v8bf*)(&sV[n * SVP + kb0 + hi * 8]);
        Bm.h[1] = *(const v8bf*)(&sV[n * SVP + kb0 + 16 + hi * 8]);
        acc = __builtin_amdgcn_wmma_f32_16x16x32_bf16(false, A.v, false, Bm.v, (short)0, acc,
                                                      false, false);
      }
#pragma unroll
      for (int j = 0; j < 8; ++j) {
        int tt = tbase + mtile * 16 + j + hi * 8;
        avb[(((size_t)b * C2_ + n) * V_ + v) * T_ + tt] = (bf16_t)acc[j];
      }
    }
  }
}

// out = (x + x1 + x2) * scale[b,c]   (x1+x2 already accumulated in out)
__global__ __launch_bounds__(256) void final_kernel(
    const float* __restrict__ x, const float* __restrict__ scale, float* __restrict__ out)
{
  size_t i = (size_t)blockIdx.x * 256 + threadIdx.x;
  int bc = (int)(i >> 13); // / (V_*T_)
  out[i] = (x[i] + out[i]) * scale[bc];
}

// ---------------------------------------------------------------------------
extern "C" void kernel_launch(void* const* d_in, const int* in_sizes, int n_in,
                              void* d_out, int out_size, void* d_ws, size_t ws_size,
                              hipStream_t stream)
{
  const float* x       = (const float*)d_in[0];
  const float* v_qw    = (const float*)d_in[1];
  const float* v_kw    = (const float*)d_in[2];
  const float* v_vw    = (const float*)d_in[3];
  const float* v_ow    = (const float*)d_in[4];
  const float* v_sigma = (const float*)d_in[5];
  const float* t_qw    = (const float*)d_in[6];
  const float* t_kw    = (const float*)d_in[7];
  const float* t_vw    = (const float*)d_in[8];
  const float* t_ow    = (const float*)d_in[9];
  const float* t_sigma = (const float*)d_in[10];
  const float* se_w1   = (const float*)d_in[11];
  const float* se_w2   = (const float*)d_in[12];
  float* out = (float*)d_out;

  const size_t NX  = (size_t)B_ * C_ * V_ * T_;
  const size_t NQK = (size_t)B_ * C8_ * V_ * T_;
  const size_t NV  = (size_t)B_ * C2_ * V_ * T_;
  const int WQK = C8_ * C_ * 9;
  const int WV  = C2_ * C_ * 9;
  const int WO  = C_ * C2_ * 9;

  bf16_t* xb   = (bf16_t*)d_ws;
  bf16_t* qb   = xb + NX;
  bf16_t* kbuf = qb + NQK;
  bf16_t* vbuf = kbuf + NQK;
  bf16_t* avb  = vbuf + NV;
  bf16_t* wq_v = avb + NV;
  bf16_t* wk_v = wq_v + WQK;
  bf16_t* wv_v = wk_v + WQK;
  bf16_t* wo_v = wv_v + WV;
  bf16_t* wq_t = wo_v + WO;
  bf16_t* wk_t = wq_t + WQK;
  bf16_t* wv_t = wk_t + WQK;
  bf16_t* wo_t = wv_t + WV;
  float* meanBuf = (float*)(wo_t + WO);
  float* seScale = meanBuf + B_ * C_;

  cvt_reduce_kernel<<<B_ * C_, 256, 0, stream>>>(x, xb, meanBuf);
  cvt_weight_kernel<<<(WQK + 255) / 256, 256, 0, stream>>>(v_qw, wq_v, C_, WQK);
  cvt_weight_kernel<<<(WQK + 255) / 256, 256, 0, stream>>>(v_kw, wk_v, C_, WQK);
  cvt_weight_kernel<<<(WV + 255) / 256, 256, 0, stream>>>(v_vw, wv_v, C_, WV);
  cvt_weight_kernel<<<(WO + 255) / 256, 256, 0, stream>>>(v_ow, wo_v, C2_, WO);
  cvt_weight_kernel<<<(WQK + 255) / 256, 256, 0, stream>>>(t_qw, wq_t, C_, WQK);
  cvt_weight_kernel<<<(WQK + 255) / 256, 256, 0, stream>>>(t_kw, wk_t, C_, WQK);
  cvt_weight_kernel<<<(WV + 255) / 256, 256, 0, stream>>>(t_vw, wv_t, C_, WV);
  cvt_weight_kernel<<<(WO + 255) / 256, 256, 0, stream>>>(t_ow, wo_t, C2_, WO);
  se_kernel<<<B_, 128, 0, stream>>>(meanBuf, se_w1, se_w2, seScale);

  conv3_bf16_kernel<C_, C8_, true ><<<dim3(2,  V_, B_), 128, 0, stream>>>(xb, wq_v, qb);
  conv3_bf16_kernel<C_, C8_, true ><<<dim3(2,  V_, B_), 128, 0, stream>>>(xb, wk_v, kbuf);
  conv3_bf16_kernel<C_, C2_, false><<<dim3(8,  V_, B_), 128, 0, stream>>>(xb, wv_v, vbuf);
  vattn_kernel<<<dim3(T_, B_), 128, 0, stream>>>(qb, kbuf, vbuf, avb);
  conv3_f32_kernel<C2_, C_, false><<<dim3(16, V_, B_), 128, 0, stream>>>(avb, wo_v, v_sigma, out);

  conv3_bf16_kernel<C_, C8_, true ><<<dim3(2,  V_, B_), 128, 0, stream>>>(xb, wq_t, qb);
  conv3_bf16_kernel<C_, C8_, true ><<<dim3(2,  V_, B_), 128, 0, stream>>>(xb, wk_t, kbuf);
  conv3_bf16_kernel<C_, C2_, false><<<dim3(8,  V_, B_), 128, 0, stream>>>(xb, wv_t, vbuf);
  tattn_kernel<<<dim3(4, V_, B_), 128, 0, stream>>>(qb, kbuf, vbuf, avb);
  conv3_f32_kernel<C2_, C_, true ><<<dim3(16, V_, B_), 128, 0, stream>>>(avb, wo_t, t_sigma, out);

  final_kernel<<<(unsigned)(NX / 256), 256, 0, stream>>>(x, seScale, out);
}
